// MessagePassingGNN_58050777972762
// MI455X (gfx1250) — matrix operand order repaired and checked by reading, not compiled, not run on hardware
//
#include <hip/hip_runtime.h>
#include <cmath>

// ---------------- problem constants ----------------
#define B_   8192
#define N_   9
#define BN_  (B_ * N_)        // 73728 nodes
#define EPER 16
#define NEDG (B_ * EPER)      // 131072 listed edges
#define ETOT (NEDG + BN_)     // 204800 edges incl self-loops (divisible by 32)
#define HD   128
#define MHD  256

// ---------------- WMMA types / helpers ----------------
typedef _Float16 v16h __attribute__((ext_vector_type(16)));
typedef _Float16 h8   __attribute__((ext_vector_type(8)));
typedef float    v8f  __attribute__((ext_vector_type(8)));

__device__ __forceinline__ v8f wmma16(v16h a, v16h b, v8f c) {
  // D = A(16x32 f16) x B(32x16 f16) + C(16x16 f32)
  return __builtin_amdgcn_wmma_f32_16x16x32_f16(false, a, false, b, (short)0, c,
                                                false, false);
}

// A fragment from a row-major f16 activation tile in LDS.
// Per CDNA5 ISA (16-bit A 16x32): lane L -> row L%16;
// halves 0..7  = K[(L/16)*8 + 0..7], halves 8..15 = K[16 + (L/16)*8 + 0..7]
__device__ __forceinline__ v16h make_a(const _Float16* act, int stride, int lane, int kt) {
  const int row = lane & 15;
  const int sub = (lane >> 4) * 8;
  const _Float16* p = act + row * stride + kt * 32;
  h8 lo = *(const h8*)(p + sub);
  h8 hi = *(const h8*)(p + 16 + sub);
  v16h a;
#pragma unroll
  for (int i = 0; i < 8; ++i) { a[i] = lo[i]; a[i + 8] = hi[i]; }
  return a;
}

// B fragment: weights pre-packed lane-major so this is one contiguous 32B load.
__device__ __forceinline__ v16h load_b16(const _Float16* Wp, int numKt, int nt, int kt,
                                         int lane) {
  return *(const v16h*)(Wp + ((size_t)((nt * numKt + kt) * 32 + lane)) * 16);
}

// C/D layout: lane L, reg v -> element [M = v + (L<16?0:8)][N = L%16]
__device__ __forceinline__ void store_tanh(_Float16* buf, int stride, v8f acc, int nt,
                                           int lane, const float* bias) {
  const int col = nt * 16 + (lane & 15);
  const float bv = bias[col];
  const int mb = (lane < 16) ? 0 : 8;
#pragma unroll
  for (int v = 0; v < 8; ++v)
    buf[(mb + v) * stride + col] = (_Float16)tanhf(acc[v] + bv);
}

__device__ __forceinline__ void atomAddF(float* p, float v) {
  __hip_atomic_fetch_add(p, v, __ATOMIC_RELAXED, __HIP_MEMORY_SCOPE_AGENT);
}

__device__ __forceinline__ float sigm(float x) { return 1.0f / (1.0f + __expf(-x)); }

// ---------------- setup kernels ----------------
__global__ void zero_kernel(float* aggr, float* deg, int nA, int nD) {
  int i = blockIdx.x * blockDim.x + threadIdx.x;
  int st = gridDim.x * blockDim.x;
  for (int j = i; j < nA; j += st) aggr[j] = 0.0f;
  for (int j = i; j < nD; j += st) deg[j] = 0.0f;
}

__global__ void edge_kernel(const int* __restrict__ edges, int* __restrict__ srcg,
                            int* __restrict__ dstg, float* __restrict__ deg) {
  int i = blockIdx.x * blockDim.x + threadIdx.x;
  if (i < NEDG) {
    int b = i >> 4, e = i & 15;
    int s = edges[b * 32 + e] + b * N_;        // edges[b,0,e]
    int d = edges[b * 32 + 16 + e] + b * N_;   // edges[b,1,e]
    srcg[i] = s;
    dstg[i] = d;
    atomAddF(&deg[d], 1.0f);
  } else if (i < ETOT) {
    int nd = i - NEDG;                          // self-loops
    srcg[i] = nd;
    dstg[i] = nd;
    atomAddF(&deg[nd], 1.0f);
  }
}

// Pack W[K][N] (f32, row-major) into WMMA-B tile order (f16).
// Assumed B layout (dense 16x16x32 f16, mirrors sparse-B doc): lane L holds
// column n = nt*16 + L%16; half h of lane L is k = kt*32 + (h/8)*16 + (L/16)*8 + h%8.
__global__ void pack_w16(const float* __restrict__ W, _Float16* __restrict__ out,
                         int K, int N) {
  int o = blockIdx.x * blockDim.x + threadIdx.x;
  if (o >= K * N) return;
  int numKt = K >> 5;
  int h = o & 15;
  int L = (o >> 4) & 31;
  int t = o >> 9;            // tile = nt*numKt + kt
  int kt = t % numKt;
  int nt = t / numKt;
  int k = kt * 32 + ((h >> 3) << 4) + ((L >> 4) << 3) + (h & 7);
  int n = (nt << 4) + (L & 15);
  out[o] = (_Float16)W[k * N + n];
}

// ---------------- encoder: x = tanh(obs @ enc_W + enc_b) ----------------
__global__ __launch_bounds__(128) void enc_kernel(const float* __restrict__ obs,
                                                  const float* __restrict__ encW,
                                                  const float* __restrict__ encB,
                                                  float* __restrict__ x) {
  int node = blockIdx.x;               // BN_ blocks
  int hcol = threadIdx.x;              // 128 threads
  int b = node / N_, n = node % N_;
  __shared__ float o_s[15];
  if (hcol < 15) o_s[hcol] = obs[b * 135 + n * 15 + hcol];
  __syncthreads();
  float acc = encB[hcol];
#pragma unroll
  for (int k = 0; k < 15; ++k) acc += o_s[k] * encW[k * HD + hcol];
  x[node * HD + hcol] = tanhf(acc);
}

// ---------------- fused per-edge message MLP + scatter (32 edges / block) ----
// Two 16-row M-tiles share every B-fragment load: 2x compute per weight fetch.
__global__ __launch_bounds__(256) void msg_kernel(
    const float* __restrict__ x, const int* __restrict__ srcg,
    const int* __restrict__ dstg, const _Float16* __restrict__ W1p,
    const float* __restrict__ b1, const _Float16* __restrict__ W2p,
    const float* __restrict__ b2, const _Float16* __restrict__ W3p,
    const float* __restrict__ b3, float* __restrict__ aggr) {
  __shared__ _Float16 bufA[32][256];   // 16 KB
  __shared__ _Float16 bufB[32][256];   // 16 KB
  __shared__ int dst_s[32];
  const int t = threadIdx.x;
  const int lane = t & 31, wave = t >> 5;
  const int eb = blockIdx.x * 32;

  if (t < 32) dst_s[t] = dstg[eb + t];
  // Fill bufA = concat(x[dst], x[src]) as f16 (cols 0..127 dst, 128..255 src).
  // 512 segments of 16 halves; each thread fills 2 segments.
#pragma unroll
  for (int sidx = t; sidx < 512; sidx += 256) {
    int r = sidx >> 4;              // edge row 0..31
    int seg = sidx & 15;            // 16-half column segment
    int node = (seg < 8) ? dstg[eb + r] : srcg[eb + r];
    const float* xp = x + (size_t)node * HD + (seg & 7) * 16;
    _Float16* dp = &bufA[r][seg * 16];
#pragma unroll
    for (int i = 0; i < 16; ++i) dp[i] = (_Float16)xp[i];
  }
  __syncthreads();

  // layer1: [32,256] x [256,256] -> tanh -> bufB
  {
    v8f a00 = {}, a01 = {}, a10 = {}, a11 = {};
    const int nt0 = wave * 2;
#pragma unroll
    for (int kt = 0; kt < 8; ++kt) {
      v16h am0 = make_a(&bufA[0][0], 256, lane, kt);    // rows 0..15
      v16h am1 = make_a(&bufA[16][0], 256, lane, kt);   // rows 16..31
      v16h bn0 = load_b16(W1p, 8, nt0, kt, lane);
      v16h bn1 = load_b16(W1p, 8, nt0 + 1, kt, lane);
      a00 = wmma16(am0, bn0, a00);
      a10 = wmma16(am1, bn0, a10);
      a01 = wmma16(am0, bn1, a01);
      a11 = wmma16(am1, bn1, a11);
    }
    store_tanh(&bufB[0][0], 256, a00, nt0, lane, b1);
    store_tanh(&bufB[0][0], 256, a01, nt0 + 1, lane, b1);
    store_tanh(&bufB[16][0], 256, a10, nt0, lane, b1);
    store_tanh(&bufB[16][0], 256, a11, nt0 + 1, lane, b1);
  }
  __syncthreads();

  // layer2: [32,256] x [256,256] -> tanh -> bufA
  {
    v8f a00 = {}, a01 = {}, a10 = {}, a11 = {};
    const int nt0 = wave * 2;
#pragma unroll
    for (int kt = 0; kt < 8; ++kt) {
      v16h am0 = make_a(&bufB[0][0], 256, lane, kt);
      v16h am1 = make_a(&bufB[16][0], 256, lane, kt);
      v16h bn0 = load_b16(W2p, 8, nt0, kt, lane);
      v16h bn1 = load_b16(W2p, 8, nt0 + 1, kt, lane);
      a00 = wmma16(am0, bn0, a00);
      a10 = wmma16(am1, bn0, a10);
      a01 = wmma16(am0, bn1, a01);
      a11 = wmma16(am1, bn1, a11);
    }
    store_tanh(&bufA[0][0], 256, a00, nt0, lane, b2);
    store_tanh(&bufA[0][0], 256, a01, nt0 + 1, lane, b2);
    store_tanh(&bufA[16][0], 256, a10, nt0, lane, b2);
    store_tanh(&bufA[16][0], 256, a11, nt0 + 1, lane, b2);
  }
  __syncthreads();

  // layer3: [32,256] x [256,128] (+b3), scatter-add into aggr[dst]
  {
    v8f ac0 = {}, ac1 = {};
#pragma unroll
    for (int kt = 0; kt < 8; ++kt) {
      v16h am0 = make_a(&bufA[0][0], 256, lane, kt);
      v16h am1 = make_a(&bufA[16][0], 256, lane, kt);
      v16h bn = load_b16(W3p, 8, wave, kt, lane);
      ac0 = wmma16(am0, bn, ac0);
      ac1 = wmma16(am1, bn, ac1);
    }
    const int col = wave * 16 + (lane & 15);
    const float bv = b3[col];
    const int mb = (lane < 16) ? 0 : 8;
#pragma unroll
    for (int v = 0; v < 8; ++v) {
      atomAddF(&aggr[(size_t)dst_s[mb + v] * HD + col], ac0[v] + bv);
      atomAddF(&aggr[(size_t)dst_s[16 + mb + v] * HD + col], ac1[v] + bv);
    }
  }
}

// ---------------- fused GRU update ----------------
__global__ __launch_bounds__(256) void gru_kernel(
    float* __restrict__ x, float* __restrict__ aggr, const float* __restrict__ deg,
    const _Float16* __restrict__ Wip, const _Float16* __restrict__ Whp,
    const float* __restrict__ bi, const float* __restrict__ bh) {
  __shared__ _Float16 aF[16][128];
  __shared__ _Float16 xF[16][128];
  __shared__ float rz_s[16][256];   // ir+hr (0..127) | iz+hz (128..255)
  __shared__ float in_s[16][128];   // inn
  __shared__ float hn_s[16][128];   // hn
  const int t = threadIdx.x;
  const int lane = t & 31, wave = t >> 5;
  const int nb = blockIdx.x * 16;

  {
    int r = t >> 4;
    int c0 = (t & 15) * 8;
    int node = nb + r;
    float dinv = 1.0f / deg[node];
#pragma unroll
    for (int i = 0; i < 8; ++i) {
      int c = c0 + i;
      float av = aggr[(size_t)node * HD + c];
      aF[r][c] = (_Float16)(av * dinv);
      aggr[(size_t)node * HD + c] = 0.0f;     // reset for next step's scatter
      xF[r][c] = (_Float16)x[(size_t)node * HD + c];
    }
  }
  __syncthreads();

  // gi = aggr@Wi + bi ; gh = x@Wh + bh ; N=384 => 24 tiles, 3 per wave
#pragma unroll
  for (int j = 0; j < 3; ++j) {
    const int nt = wave * 3 + j;
    v8f gi = {}, gh = {};
#pragma unroll
    for (int kt = 0; kt < 4; ++kt) {
      v16h aa = make_a(&aF[0][0], 128, lane, kt);
      v16h xa = make_a(&xF[0][0], 128, lane, kt);
      gi = wmma16(aa, load_b16(Wip, 4, nt, kt, lane), gi);
      gh = wmma16(xa, load_b16(Whp, 4, nt, kt, lane), gh);
    }
    const int col = nt * 16 + (lane & 15);
    const float bvi = bi[col], bvh = bh[col];
    const int mb = (lane < 16) ? 0 : 8;
    if (col < 256) {              // uniform per tile (256 is tile-aligned)
#pragma unroll
      for (int v = 0; v < 8; ++v)
        rz_s[mb + v][col] = (gi[v] + bvi) + (gh[v] + bvh);
    } else {
      int c2 = col - 256;
#pragma unroll
      for (int v = 0; v < 8; ++v) {
        in_s[mb + v][c2] = gi[v] + bvi;
        hn_s[mb + v][c2] = gh[v] + bvh;
      }
    }
  }
  __syncthreads();

  {
    int r = t >> 4;
    int c0 = (t & 15) * 8;
    int node = nb + r;
#pragma unroll
    for (int i = 0; i < 8; ++i) {
      int c = c0 + i;
      float rr = sigm(rz_s[r][c]);
      float zz = sigm(rz_s[r][c + 128]);
      float nn = tanhf(in_s[r][c] + rr * hn_s[r][c]);
      float xo = x[(size_t)node * HD + c];
      x[(size_t)node * HD + c] = (1.0f - zz) * nn + zz * xo;
    }
  }
}

// ---------------- fused decoder ----------------
__global__ __launch_bounds__(256) void dec_kernel(
    const float* __restrict__ x, const _Float16* __restrict__ W1p,
    const float* __restrict__ b1, const _Float16* __restrict__ W2p,
    const float* __restrict__ b2, const float* __restrict__ W3,
    const float* __restrict__ b3, float* __restrict__ out) {
  __shared__ _Float16 xF[16][128];
  __shared__ _Float16 d1[16][256];
  __shared__ _Float16 d2[16][256];
  const int t = threadIdx.x;
  const int lane = t & 31, wave = t >> 5;
  const int nb = blockIdx.x * 16;

  {
    int r = t >> 4;
    int c0 = (t & 15) * 8;
    int node = nb + r;
#pragma unroll
    for (int i = 0; i < 8; ++i)
      xF[r][c0 + i] = (_Float16)x[(size_t)node * HD + c0 + i];
  }
  __syncthreads();

  // layer1: [16,128] x [128,256] -> tanh -> d1
  {
    v8f a0 = {}, a1 = {};
    const int nt0 = wave * 2;
#pragma unroll
    for (int kt = 0; kt < 4; ++kt) {
      v16h a = make_a(&xF[0][0], 128, lane, kt);
      a0 = wmma16(a, load_b16(W1p, 4, nt0, kt, lane), a0);
      a1 = wmma16(a, load_b16(W1p, 4, nt0 + 1, kt, lane), a1);
    }
    store_tanh(&d1[0][0], 256, a0, nt0, lane, b1);
    store_tanh(&d1[0][0], 256, a1, nt0 + 1, lane, b1);
  }
  __syncthreads();

  // layer2: [16,256] x [256,256] -> tanh -> d2
  {
    v8f a0 = {}, a1 = {};
    const int nt0 = wave * 2;
#pragma unroll
    for (int kt = 0; kt < 8; ++kt) {
      v16h a = make_a(&d1[0][0], 256, lane, kt);
      a0 = wmma16(a, load_b16(W2p, 8, nt0, kt, lane), a0);
      a1 = wmma16(a, load_b16(W2p, 8, nt0 + 1, kt, lane), a1);
    }
    store_tanh(&d2[0][0], 256, a0, nt0, lane, b2);
    store_tanh(&d2[0][0], 256, a1, nt0 + 1, lane, b2);
  }
  __syncthreads();

  // layer3: [16,256] x [256,1]; masked write out[b, n<8]
  if (t < 16) {
    int node = nb + t;
    float acc = b3[0];
#pragma unroll 8
    for (int k = 0; k < 256; ++k) acc += (float)d2[t][k] * W3[k];
    int b = node / N_, n = node % N_;
    if (n < 8) out[(size_t)b * 8 + n] = acc;
  }
}

// ---------------- host launcher ----------------
extern "C" void kernel_launch(void* const* d_in, const int* in_sizes, int n_in,
                              void* d_out, int out_size, void* d_ws, size_t ws_size,
                              hipStream_t stream) {
  const float* obs  = (const float*)d_in[0];
  const int* edges  = (const int*)d_in[1];
  const float* encW = (const float*)d_in[2];
  const float* encB = (const float*)d_in[3];
  const float* mW1  = (const float*)d_in[4];
  const float* mB1  = (const float*)d_in[5];
  const float* mW2  = (const float*)d_in[6];
  const float* mB2  = (const float*)d_in[7];
  const float* mW3  = (const float*)d_in[8];
  const float* mB3  = (const float*)d_in[9];
  const float* gWi  = (const float*)d_in[10];
  const float* gWh  = (const float*)d_in[11];
  const float* gBi  = (const float*)d_in[12];
  const float* gBh  = (const float*)d_in[13];
  const float* dW1  = (const float*)d_in[14];
  const float* dB1  = (const float*)d_in[15];
  const float* dW2  = (const float*)d_in[16];
  const float* dB2  = (const float*)d_in[17];
  const float* dW3  = (const float*)d_in[18];
  const float* dB3  = (const float*)d_in[19];
  float* out = (float*)d_out;

  char* ws = (char*)d_ws;
  size_t off = 0;
  auto take = [&](size_t bytes) -> void* {
    void* p = ws + off;
    off = (off + bytes + 255) & ~(size_t)255;
    return p;
  };
  float* x     = (float*)take((size_t)BN_ * HD * 4);
  float* aggr  = (float*)take((size_t)BN_ * HD * 4);
  float* deg   = (float*)take((size_t)BN_ * 4);
  int* srcg    = (int*)take((size_t)ETOT * 4);
  int* dstg    = (int*)take((size_t)ETOT * 4);
  _Float16* W1p  = (_Float16*)take((size_t)4 * 256 * 256 * 2);
  _Float16* W2p  = (_Float16*)take((size_t)4 * 256 * 256 * 2);
  _Float16* W3p  = (_Float16*)take((size_t)4 * 256 * 128 * 2);
  _Float16* Wip  = (_Float16*)take((size_t)4 * 128 * 384 * 2);
  _Float16* Whp  = (_Float16*)take((size_t)4 * 128 * 384 * 2);
  _Float16* dW1p = (_Float16*)take((size_t)128 * 256 * 2);
  _Float16* dW2p = (_Float16*)take((size_t)256 * 256 * 2);
  (void)in_sizes; (void)n_in; (void)out_size; (void)ws_size;

  zero_kernel<<<2048, 256, 0, stream>>>(aggr, deg, BN_ * HD, BN_);
  edge_kernel<<<(ETOT + 255) / 256, 256, 0, stream>>>(edges, srcg, dstg, deg);

  for (int s = 0; s < 4; ++s) {
    pack_w16<<<(256 * 256 + 255) / 256, 256, 0, stream>>>(mW1 + s * 256 * 256,
                                                          W1p + s * 256 * 256, 256, 256);
    pack_w16<<<(256 * 256 + 255) / 256, 256, 0, stream>>>(mW2 + s * 256 * 256,
                                                          W2p + s * 256 * 256, 256, 256);
    pack_w16<<<(256 * 128 + 255) / 256, 256, 0, stream>>>(mW3 + s * 256 * 128,
                                                          W3p + s * 256 * 128, 256, 128);
    pack_w16<<<(128 * 384 + 255) / 256, 256, 0, stream>>>(gWi + s * 128 * 384,
                                                          Wip + s * 128 * 384, 128, 384);
    pack_w16<<<(128 * 384 + 255) / 256, 256, 0, stream>>>(gWh + s * 128 * 384,
                                                          Whp + s * 128 * 384, 128, 384);
  }
  pack_w16<<<(128 * 256 + 255) / 256, 256, 0, stream>>>(dW1, dW1p, 128, 256);
  pack_w16<<<(256 * 256 + 255) / 256, 256, 0, stream>>>(dW2, dW2p, 256, 256);

  enc_kernel<<<BN_, 128, 0, stream>>>(obs, encW, encB, x);

  for (int s = 0; s < 4; ++s) {
    msg_kernel<<<ETOT / 32, 256, 0, stream>>>(
        x, srcg, dstg, W1p + s * 256 * 256, mB1 + s * 256, W2p + s * 256 * 256,
        mB2 + s * 256, W3p + s * 256 * 128, mB3 + s * 128, aggr);
    gru_kernel<<<BN_ / 16, 256, 0, stream>>>(x, aggr, deg, Wip + s * 128 * 384,
                                             Whp + s * 128 * 384, gBi + s * 384,
                                             gBh + s * 384);
  }

  dec_kernel<<<BN_ / 16, 256, 0, stream>>>(x, dW1p, dB1, dW2p, dB2, dW3, dB3, out);
}
